// Decoder_54013508714698
// MI455X (gfx1250) — matrix operand order repaired and checked by reading, not compiled, not run on hardware
//
#include <hip/hip_runtime.h>

typedef __attribute__((ext_vector_type(16))) __bf16 v16bf;
typedef __attribute__((ext_vector_type(8)))  __bf16 v8bf;
typedef __attribute__((ext_vector_type(2)))  __bf16 v2bf;
typedef __attribute__((ext_vector_type(8)))  float  v8f;
typedef __attribute__((ext_vector_type(4)))  float  v4f;

#define DMODEL 512
#define NHEAD  8
#define HDIM   64
#define BATCH  64
#define NSEQ   2048
#define NEGINF (-1e15f)

#define SHUF16(lo, hi) __builtin_shufflevector(lo, hi, \
    0,1,2,3,4,5,6,7,8,9,10,11,12,13,14,15)

// ---------------------------------------------------------------------------
// Phase 0: convert Wk / Wv to bf16 (weights then live in L2 for the big GEMM)
// ---------------------------------------------------------------------------
__global__ void cvt_w_kernel(const float* __restrict__ Wk,
                             const float* __restrict__ Wv,
                             __bf16* __restrict__ Wkb,
                             __bf16* __restrict__ Wvb) {
    int i = blockIdx.x * 256 + threadIdx.x;
    if (i < DMODEL * DMODEL) {
        Wkb[i] = (__bf16)Wk[i];
        Wvb[i] = (__bf16)Wv[i];
    }
}

// ---------------------------------------------------------------------------
// Phase 1: K = x@Wk+bk, V = x@Wv+bv  (fused; x staged once per 32-row tile).
// 256 threads = 8 waves; waves 0-3 -> K, 4-7 -> V; two 64-column passes per
// wave (2 row-tiles x 4 col-tiles, 8 v8f accumulators). k-loop is manually
// 2x unrolled with ping-pong A/B fragment buffers: every WMMA consumes
// fragments loaded one half-iteration earlier. The final prefetch runs 32
// rows past the weight table (lands in our own workspace, value discarded)
// so all addresses stay affine -- no per-iteration address recompute.
// Epilogue stages D-fragments through LDS for coalesced b128 stores.
// ---------------------------------------------------------------------------
__global__ __launch_bounds__(256)
void gemm_kv_kernel(const float*  __restrict__ x,
                    const __bf16* __restrict__ Wkb,
                    const __bf16* __restrict__ Wvb,
                    const float*  __restrict__ bk,
                    const float*  __restrict__ bv,
                    __bf16* __restrict__ Kb,
                    __bf16* __restrict__ Vb) {
    __shared__ __bf16 xs[32][520];   // 33,280 B: padded stride, conflict-free b128 A reads
    __shared__ __bf16 os[32][264];   // 16,896 B: output staging for coalesced stores

    const int r0  = blockIdx.x * 32;         // global row base (row = b*NSEQ + n)
    const int tid = threadIdx.x;

    // Stage 32x512 fp32 tile -> bf16 LDS (nontemporal float4 loads: x is
    // touched exactly once grid-wide, keep it out of L2's resident set)
    #pragma unroll
    for (int it = 0; it < 16; ++it) {
        int idx = it * 256 + tid;            // float4 index in tile (4096 total)
        int r   = idx >> 7;                  // 128 float4 per row
        int c   = (idx & 127) << 2;
        v4f f = __builtin_nontemporal_load(
                    (const v4f*)(x + (size_t)(r0 + r) * DMODEL + c));
        xs[r][c + 0] = (__bf16)f[0];
        xs[r][c + 1] = (__bf16)f[1];
        xs[r][c + 2] = (__bf16)f[2];
        xs[r][c + 3] = (__bf16)f[3];
    }
    __syncthreads();

    const int wv   = tid >> 5;               // wave id 0..7
    const int lane = tid & 31;
    const int mat  = wv >> 2;                // 0 = K, 1 = V
    const int slot = (wv & 3) * 64;          // staging column base for this wave
    const __bf16* Wb   = mat ? Wvb : Wkb;
    const float*  bias = mat ? bv  : bk;

    // ISA 16-bit A layout: lanes 0-15 row M hold K {0..7,16..23},
    // lanes 16-31 hold K {8..15,24..31}
    const int mrow = lane & 15;
    const int klo  = (lane < 16) ? 0 : 8;
    const int rsub = (lane < 16) ? 0 : 8;

    for (int p = 0; p < 2; ++p) {
        const int gcol0 = (wv & 3) * 128 + p * 64;   // wave's global column base

        v8f acc[2][4];
        #pragma unroll
        for (int i = 0; i < 2; ++i)
            #pragma unroll
            for (int j = 0; j < 4; ++j)
                acc[i][j] = (v8f){0.f,0.f,0.f,0.f,0.f,0.f,0.f,0.f};

        // Ping-pong fragment buffers. B layout: lane&15 = N column, elements
        // over K in the same order as A ({0..7,16..23} / {8..15,24..31}).
        v16bf Ba[4], Bb[4], A0a, A1a, A0b, A1b;

        // Preload k-step 0 into buffer 'a'
        {
            const __bf16* wrow = Wb + (size_t)klo * DMODEL + gcol0 + mrow;
            #pragma unroll
            for (int ct = 0; ct < 4; ++ct)
                #pragma unroll
                for (int s = 0; s < 16; ++s)
                    Ba[ct][s] = wrow[(size_t)((s & 7) + ((s >> 3) << 4)) * DMODEL + ct * 16];
            v8bf l0 = *(const v8bf*)&xs[mrow     ][klo];
            v8bf h0 = *(const v8bf*)&xs[mrow     ][klo + 16];
            v8bf l1 = *(const v8bf*)&xs[16 + mrow][klo];
            v8bf h1 = *(const v8bf*)&xs[16 + mrow][klo + 16];
            A0a = SHUF16(l0, h0);
            A1a = SHUF16(l1, h1);
        }

        for (int ks = 0; ks < DMODEL; ks += 64) {
            // ---- half 1: prefetch ks+32 into 'b', compute with 'a' ----
            {
                const int kn = ks + 32;                       // always < DMODEL
                const __bf16* wrow = Wb + (size_t)(kn + klo) * DMODEL + gcol0 + mrow;
                #pragma unroll
                for (int ct = 0; ct < 4; ++ct)
                    #pragma unroll
                    for (int s = 0; s < 16; ++s)
                        Bb[ct][s] = wrow[(size_t)((s & 7) + ((s >> 3) << 4)) * DMODEL + ct * 16];
                v8bf l0 = *(const v8bf*)&xs[mrow     ][kn + klo];
                v8bf h0 = *(const v8bf*)&xs[mrow     ][kn + klo + 16];
                v8bf l1 = *(const v8bf*)&xs[16 + mrow][kn + klo];
                v8bf h1 = *(const v8bf*)&xs[16 + mrow][kn + klo + 16];
                A0b = SHUF16(l0, h0);
                A1b = SHUF16(l1, h1);

                #pragma unroll
                for (int ct = 0; ct < 4; ++ct) {
                    acc[0][ct] = __builtin_amdgcn_wmma_f32_16x16x32_bf16(
                        false, A0a, false, Ba[ct], (short)0, acc[0][ct], false, false);
                    acc[1][ct] = __builtin_amdgcn_wmma_f32_16x16x32_bf16(
                        false, A1a, false, Ba[ct], (short)0, acc[1][ct], false, false);
                }
            }
            // ---- half 2: prefetch ks+64 into 'a', compute with 'b' ----
            // Final prefetch (ks+64 == DMODEL) reads 32 rows past the weight
            // table into adjacent workspace; value never consumed. LDS index
            // is wrapped (cheap AND) to stay in-bounds.
            {
                const int kn  = ks + 64;
                const int knl = kn & (DMODEL - 1);
                const __bf16* wrow = Wb + (size_t)(kn + klo) * DMODEL + gcol0 + mrow;
                #pragma unroll
                for (int ct = 0; ct < 4; ++ct)
                    #pragma unroll
                    for (int s = 0; s < 16; ++s)
                        Ba[ct][s] = wrow[(size_t)((s & 7) + ((s >> 3) << 4)) * DMODEL + ct * 16];
                v8bf l0 = *(const v8bf*)&xs[mrow     ][knl + klo];
                v8bf h0 = *(const v8bf*)&xs[mrow     ][knl + klo + 16];
                v8bf l1 = *(const v8bf*)&xs[16 + mrow][knl + klo];
                v8bf h1 = *(const v8bf*)&xs[16 + mrow][knl + klo + 16];
                A0a = SHUF16(l0, h0);
                A1a = SHUF16(l1, h1);

                #pragma unroll
                for (int ct = 0; ct < 4; ++ct) {
                    acc[0][ct] = __builtin_amdgcn_wmma_f32_16x16x32_bf16(
                        false, A0b, false, Bb[ct], (short)0, acc[0][ct], false, false);
                    acc[1][ct] = __builtin_amdgcn_wmma_f32_16x16x32_bf16(
                        false, A1b, false, Bb[ct], (short)0, acc[1][ct], false, false);
                }
            }
        }

        // ---- epilogue: stage to LDS, then coalesced b128 stores ----
        // C/D layout: lane<16 -> M=j, lane>=16 -> M=8+j; N = lane&15.
        __syncthreads();                      // prior os readers done
        if (mat == 0) {
            #pragma unroll
            for (int ct = 0; ct < 4; ++ct) {
                int   scol = slot + ct * 16 + (lane & 15);
                float bc   = bias[gcol0 + ct * 16 + (lane & 15)];
                #pragma unroll
                for (int rt = 0; rt < 2; ++rt)
                    #pragma unroll
                    for (int j = 0; j < 8; ++j)
                        os[rt * 16 + rsub + j][scol] = (__bf16)(acc[rt][ct][j] + bc);
            }
        }
        __syncthreads();
        #pragma unroll
        for (int i2 = 0; i2 < 4; ++i2) {      // 1024 b128 chunks / 256 threads
            int i  = i2 * 256 + tid;
            int r  = i >> 5;                  // 32 chunks per row
            int cc = (i & 31) << 3;           // staging col (multiple of 8)
            int gc = ((cc >> 6) << 7) + p * 64 + (cc & 63);
            *(v8bf*)(Kb + (size_t)(r0 + r) * DMODEL + gc) = *(const v8bf*)&os[r][cc];
        }
        __syncthreads();
        if (mat == 1) {
            #pragma unroll
            for (int ct = 0; ct < 4; ++ct) {
                int   scol = slot + ct * 16 + (lane & 15);
                float bc   = bias[gcol0 + ct * 16 + (lane & 15)];
                #pragma unroll
                for (int rt = 0; rt < 2; ++rt)
                    #pragma unroll
                    for (int j = 0; j < 8; ++j)
                        os[rt * 16 + rsub + j][scol] = (__bf16)(acc[rt][ct][j] + bc);
            }
        }
        __syncthreads();
        #pragma unroll
        for (int i2 = 0; i2 < 4; ++i2) {
            int i  = i2 * 256 + tid;
            int r  = i >> 5;
            int cc = (i & 31) << 3;
            int gc = ((cc >> 6) << 7) + p * 64 + (cc & 63);
            *(v8bf*)(Vb + (size_t)(r0 + r) * DMODEL + gc) = *(const v8bf*)&os[r][cc];
        }
    }
}

// ---------------------------------------------------------------------------
// Phase 2: Q = [h_g | v1 | v2] @ Wq + bq   (tiny; coalesced over out columns)
// ---------------------------------------------------------------------------
__global__ __launch_bounds__(256)
void proj_q_kernel(const float* __restrict__ hg, const float* __restrict__ v1,
                   const float* __restrict__ v2, const float* __restrict__ Wq,
                   const float* __restrict__ bq, float* __restrict__ Q) {
    int g = blockIdx.x * 256 + threadIdx.x;      // BATCH*DMODEL threads
    int o = g & (DMODEL - 1);
    int b = g >> 9;
    float s = bq[o];
    const float* w  = Wq + o;
    const float* hb = hg + b * DMODEL;
    for (int k = 0; k < DMODEL; ++k) s += hb[k] * w[k * DMODEL];
    w += DMODEL * DMODEL;
    for (int k = 0; k < DMODEL; ++k) s += v1[k] * w[k * DMODEL];
    w += DMODEL * DMODEL;
    for (int k = 0; k < DMODEL; ++k) s += v2[k] * w[k * DMODEL];
    Q[g] = s;
}

// ---------------------------------------------------------------------------
// Phase 3: single-query attention per (b,h). b128 K reads, block softmax,
// weighted-V accumulation with b32 (2x bf16) loads.
// ---------------------------------------------------------------------------
__global__ __launch_bounds__(256)
void attn_kernel(const __bf16* __restrict__ Kb, const __bf16* __restrict__ Vb,
                 const float* __restrict__ Q, const int* __restrict__ mask,
                 float* __restrict__ u) {
    __shared__ float sc[NSEQ];
    __shared__ float qv[HDIM];
    __shared__ float red[512];

    const int b   = blockIdx.x >> 3;
    const int h   = blockIdx.x & 7;
    const int tid = threadIdx.x;

    if (tid < HDIM) qv[tid] = Q[b * DMODEL + h * HDIM + tid];
    __syncthreads();

    for (int n = tid; n < NSEQ; n += 256) {
        const v8bf* kp = (const v8bf*)(Kb + ((size_t)(b * NSEQ + n)) * DMODEL + h * HDIM);
        float d = 0.f;
        #pragma unroll
        for (int c = 0; c < 8; ++c) {
            v8bf kk = kp[c];
            #pragma unroll
            for (int j = 0; j < 8; ++j) d += qv[c * 8 + j] * (float)kk[j];
        }
        d *= 0.125f;                                   // 1/sqrt(64)
        if (mask[b * NSEQ + n] == 0) d = NEGINF;
        sc[n] = d;
    }
    __syncthreads();

    float m = -3.4e38f;
    for (int n = tid; n < NSEQ; n += 256) m = fmaxf(m, sc[n]);
    red[tid] = m; __syncthreads();
    for (int s = 128; s > 0; s >>= 1) {
        if (tid < s) red[tid] = fmaxf(red[tid], red[tid + s]);
        __syncthreads();
    }
    m = red[0]; __syncthreads();

    float lsum = 0.f;
    for (int n = tid; n < NSEQ; n += 256) {
        float e = __expf(sc[n] - m);
        sc[n] = e;
        lsum += e;
    }
    red[tid] = lsum; __syncthreads();
    for (int s = 128; s > 0; s >>= 1) {
        if (tid < s) red[tid] += red[tid + s];
        __syncthreads();
    }
    float inv = 1.0f / red[0];
    __syncthreads();

    // u[d] = sum_n attn[n] * V[n,d] ; 8 n-groups, 2 d's per thread (b32 loads)
    const int d2  = (tid & 31) * 2;
    const int grp = tid >> 5;                          // 0..7
    float a0 = 0.f, a1 = 0.f;
    for (int n = grp; n < NSEQ; n += 8) {
        v2bf vv = *(const v2bf*)(Vb + ((size_t)(b * NSEQ + n)) * DMODEL + h * HDIM + d2);
        float w = sc[n];
        a0 += w * (float)vv[0];
        a1 += w * (float)vv[1];
    }
    red[grp * 64 + d2]     = a0;
    red[grp * 64 + d2 + 1] = a1;
    __syncthreads();
    if (tid < HDIM) {
        float t = 0.f;
        #pragma unroll
        for (int g2 = 0; g2 < 8; ++g2) t += red[g2 * 64 + tid];
        u[b * DMODEL + h * HDIM + tid] = t * inv;
    }
}

// ---------------------------------------------------------------------------
// Phase 4: glimpse projection g = u @ Wh + bh
// ---------------------------------------------------------------------------
__global__ __launch_bounds__(256)
void proj_g_kernel(const float* __restrict__ u, const float* __restrict__ Wh,
                   const float* __restrict__ bh, float* __restrict__ g) {
    int gi = blockIdx.x * 256 + threadIdx.x;
    int o = gi & (DMODEL - 1);
    int b = gi >> 9;
    float s = bh[o];
    const float* ub = u + b * DMODEL;
    for (int k = 0; k < DMODEL; ++k) s += ub[k] * Wh[k * DMODEL + o];
    g[gi] = s;
}

// ---------------------------------------------------------------------------
// Phase 5a: pointer logits = 10*tan(g . K[n]) with mask, into ws (fp32).
// 8 threads cooperate per row with interleaved chunks -> 128 B contiguous
// per group per step; 8-wide shfl reduction (groups are 8-aligned in wave32).
// ---------------------------------------------------------------------------
__global__ __launch_bounds__(256)
void pointer_logits_kernel(const __bf16* __restrict__ Kb,
                           const float* __restrict__ g,
                           const int* __restrict__ mask,
                           float* __restrict__ LG) {
    __shared__ float gv[DMODEL];
    const int tid  = threadIdx.x;
    const int b    = blockIdx.x >> 6;        // 64 blocks per batch
    const int rloc = tid >> 3;               // 32 rows per block
    const int j    = tid & 7;                // 8 threads per row
    const int n    = (blockIdx.x & 63) * 32 + rloc;

    for (int i = tid; i < DMODEL; i += 256) gv[i] = g[b * DMODEL + i];
    __syncthreads();

    const v8bf* kp = (const v8bf*)(Kb + ((size_t)(b * NSEQ + n)) * DMODEL);
    float d = 0.f;
    #pragma unroll
    for (int c = 0; c < 8; ++c) {
        int  ch = c * 8 + j;                 // interleaved -> contiguous per group
        v8bf kk = kp[ch];
        #pragma unroll
        for (int e = 0; e < 8; ++e) d += gv[ch * 8 + e] * (float)kk[e];
    }
    #pragma unroll
    for (int off = 4; off > 0; off >>= 1) d += __shfl_down(d, off, 8);

    if (j == 0) {
        float v = 10.0f * __tanf(d);
        if (mask[b * NSEQ + n] == 0) v = NEGINF;
        LG[b * NSEQ + n] = v;
    }
}

// ---------------------------------------------------------------------------
// Phase 5b: per-batch masked softmax + first-occurrence argmax over logits.
// Writes (float)index to out[0..63] and prob to out[64..127].
// ---------------------------------------------------------------------------
__global__ __launch_bounds__(256)
void pointer_reduce_kernel(const float* __restrict__ LG, float* __restrict__ out) {
    __shared__ float lg[NSEQ];
    __shared__ float rv[256];
    __shared__ int   ri[256];

    const int b   = blockIdx.x;
    const int tid = threadIdx.x;

    for (int n = tid; n < NSEQ; n += 256) lg[n] = LG[b * NSEQ + n];
    __syncthreads();

    // argmax with first-occurrence tie-break
    float bm = -3.4e38f; int bi = NSEQ;
    for (int n = tid; n < NSEQ; n += 256) {
        float v = lg[n];
        if (v > bm) { bm = v; bi = n; }
    }
    rv[tid] = bm; ri[tid] = bi; __syncthreads();
    for (int s = 128; s > 0; s >>= 1) {
        if (tid < s) {
            float vo = rv[tid + s]; int io = ri[tid + s];
            if (vo > rv[tid] || (vo == rv[tid] && io < ri[tid])) {
                rv[tid] = vo; ri[tid] = io;
            }
        }
        __syncthreads();
    }
    float mx = rv[0]; int idx = ri[0];
    __syncthreads();

    float ls = 0.f;
    for (int n = tid; n < NSEQ; n += 256) ls += __expf(lg[n] - mx);
    rv[tid] = ls; __syncthreads();
    for (int s = 128; s > 0; s >>= 1) {
        if (tid < s) rv[tid] += rv[tid + s];
        __syncthreads();
    }
    if (tid == 0) {
        out[b]         = (float)idx;                       // vertexes
        out[BATCH + b] = __expf(lg[idx] - mx) / rv[0];     // probs
    }
}

// ---------------------------------------------------------------------------
extern "C" void kernel_launch(void* const* d_in, const int* in_sizes, int n_in,
                              void* d_out, int out_size, void* d_ws, size_t ws_size,
                              hipStream_t stream) {
    const float* x    = (const float*)d_in[0];
    const float* h_g  = (const float*)d_in[1];
    const int*   mask = (const int*)  d_in[2];
    const float* v1   = (const float*)d_in[3];
    const float* v2   = (const float*)d_in[4];
    const float* Wk   = (const float*)d_in[5];
    const float* bk   = (const float*)d_in[6];
    const float* Wv   = (const float*)d_in[7];
    const float* bv   = (const float*)d_in[8];
    const float* Wq   = (const float*)d_in[9];
    const float* bq   = (const float*)d_in[10];
    const float* Wh   = (const float*)d_in[11];
    const float* bh   = (const float*)d_in[12];

    char*  ws  = (char*)d_ws;
    size_t off = 0;
    __bf16* Kb  = (__bf16*)(ws + off); off += (size_t)BATCH * NSEQ * DMODEL * 2;
    __bf16* Vb  = (__bf16*)(ws + off); off += (size_t)BATCH * NSEQ * DMODEL * 2;
    __bf16* Wkb = (__bf16*)(ws + off); off += (size_t)DMODEL * DMODEL * 2;
    __bf16* Wvb = (__bf16*)(ws + off); off += (size_t)DMODEL * DMODEL * 2;
    float*  Q   = (float*)(ws + off);  off += (size_t)BATCH * DMODEL * 4;   // (also overrun pad)
    float*  U   = (float*)(ws + off);  off += (size_t)BATCH * DMODEL * 4;
    float*  G   = (float*)(ws + off);  off += (size_t)BATCH * DMODEL * 4;
    float*  LG  = (float*)(ws + off);  off += (size_t)BATCH * NSEQ * 4;
    (void)ws_size; (void)in_sizes; (void)n_in; (void)out_size;

    cvt_w_kernel          <<<(DMODEL * DMODEL) / 256, 256, 0, stream>>>(Wk, Wv, Wkb, Wvb);
    gemm_kv_kernel        <<<(BATCH * NSEQ) / 32,     256, 0, stream>>>(x, Wkb, Wvb, bk, bv, Kb, Vb);
    proj_q_kernel         <<<(BATCH * DMODEL) / 256,  256, 0, stream>>>(h_g, v1, v2, Wq, bq, Q);
    attn_kernel           <<<BATCH * NHEAD,           256, 0, stream>>>(Kb, Vb, Q, mask, U);
    proj_g_kernel         <<<(BATCH * DMODEL) / 256,  256, 0, stream>>>(U, Wh, bh, G);
    pointer_logits_kernel <<<BATCH * 64,              256, 0, stream>>>(Kb, G, mask, LG);
    pointer_reduce_kernel <<<BATCH,                   256, 0, stream>>>(LG, (float*)d_out);
}
